// MaskAttention_55490977465230
// MI455X (gfx1250) — matrix operand order repaired and checked by reading, not compiled, not run on hardware
//
#include <hip/hip_runtime.h>
#include <hip/hip_bf16.h>

// FlashAttention (causal) for B=2, L=2048, H=16, E=64, fp32 in/out.
// Pre-pass: fp32->f16 convert (Q scaled, K) + V transpose into d_ws.
// Main: f16 WMMA for both GEMMs, f32 online softmax, double-buffered
// async-to-LDS tile staging (CDNA5 ASYNCcnt path), row-sum via WMMA-with-ones,
// row-max via v_permlane16 butterflies.

typedef __attribute__((ext_vector_type(16))) _Float16 v16h;
typedef __attribute__((ext_vector_type(8)))  _Float16 v8h;
typedef __attribute__((ext_vector_type(4)))  _Float16 v4h;
typedef __attribute__((ext_vector_type(8)))  float    v8f;
typedef __attribute__((ext_vector_type(4)))  int      v4i_t;

#define ATT_B 2
#define ATT_L 2048
#define ATT_H 16
#define ATT_E 64
#define LDK   72   // LDS row stride in halves (64 + 8 pad); 144B = 9*16B -> b128-aligned

// ---------------- helpers ----------------

__device__ __forceinline__ v16h lds_read_2x8(const _Float16* p0, const _Float16* p1) {
    v8h a = *(const v8h*)p0;
    v8h b = *(const v8h*)p1;
    v16h r;
#pragma unroll
    for (int i = 0; i < 8; ++i) { r[i] = a[i]; r[i + 8] = b[i]; }
    return r;
}
__device__ __forceinline__ v16h lds_read16(const _Float16* p) {
    return lds_read_2x8(p, p + 8);
}

__device__ __forceinline__ v8f wmma_f16(v16h a, v16h b, v8f c) {
    return __builtin_amdgcn_wmma_f32_16x16x32_f16(false, a, false, b, (short)0, c, false, false);
}

// 16-byte global -> LDS copy on the CDNA5 async path when available.
__device__ __forceinline__ void async_copy16(const _Float16* g, _Float16* l) {
#if __has_builtin(__builtin_amdgcn_global_load_async_to_lds_b128)
    __builtin_amdgcn_global_load_async_to_lds_b128(
        (v4i_t*)g,
        (__attribute__((address_space(3))) v4i_t*)l,
        0, 0);
#else
    *(v8h*)l = *(const v8h*)g;
#endif
}

__device__ __forceinline__ void wait_async0() {
#if __has_builtin(__builtin_amdgcn_s_wait_asynccnt)
    __builtin_amdgcn_s_wait_asynccnt(0);
#else
    asm volatile("s_wait_asynccnt 0" ::: "memory");
#endif
}

// max-butterfly across the 16-lane N group (VALU crossbar when available)
__device__ __forceinline__ float xor16_max(float x) {
#if __has_builtin(__builtin_amdgcn_permlane16)
    int v = __float_as_int(x);
    int t;
    t = __builtin_amdgcn_permlane16(v, v, 0x67452301u, 0xEFCDAB89u, false, false); // xor 1
    v = __float_as_int(fmaxf(__int_as_float(v), __int_as_float(t)));
    t = __builtin_amdgcn_permlane16(v, v, 0x54761032u, 0xDCFE98BAu, false, false); // xor 2
    v = __float_as_int(fmaxf(__int_as_float(v), __int_as_float(t)));
    t = __builtin_amdgcn_permlane16(v, v, 0x32107654u, 0xBA98FEDCu, false, false); // xor 4
    v = __float_as_int(fmaxf(__int_as_float(v), __int_as_float(t)));
    t = __builtin_amdgcn_permlane16(v, v, 0xFEDCBA98u, 0x76543210u, false, false); // xor 8
    return fmaxf(__int_as_float(v), __int_as_float(t));
#else
    x = fmaxf(x, __shfl_xor(x, 1, 16));
    x = fmaxf(x, __shfl_xor(x, 2, 16));
    x = fmaxf(x, __shfl_xor(x, 4, 16));
    x = fmaxf(x, __shfl_xor(x, 8, 16));
    return x;
#endif
}

// ---------------- pre-pass kernels ----------------

// fp32 -> f16 for Q (scale folded) and K, layout preserved [B,L,H,E]
__global__ void __launch_bounds__(256)
cvt_qk_kernel(const float* __restrict__ Q, const float* __restrict__ K,
              _Float16* __restrict__ Qh, _Float16* __restrict__ Kh)
{
    const int idx = blockIdx.x * 256 + threadIdx.x;
#pragma unroll
    for (int j = 0; j < 2; ++j) {
        const size_t i4 = (size_t)idx * 2 + j;       // float4 index
        const float4 q = *(const float4*)(Q + i4 * 4);
        v4h hq = { (_Float16)(q.x * 0.125f), (_Float16)(q.y * 0.125f),
                   (_Float16)(q.z * 0.125f), (_Float16)(q.w * 0.125f) };
        *(v4h*)(Qh + i4 * 4) = hq;
        const float4 k = *(const float4*)(K + i4 * 4);
        v4h hk = { (_Float16)k.x, (_Float16)k.y, (_Float16)k.z, (_Float16)k.w };
        *(v4h*)(Kh + i4 * 4) = hk;
    }
}

// V [B,L,H,E] fp32 -> Vt [B,H,E,L] f16 (LDS-tiled transpose, 64x64 tiles)
__global__ void __launch_bounds__(128)
transpose_v_kernel(const float* __restrict__ V, _Float16* __restrict__ Vt)
{
    __shared__ _Float16 t[64 * LDK];
    const int tid = threadIdx.x;
    const int nq = ATT_L / 64;
    int idx = blockIdx.x;
    const int b  = idx / (ATT_H * nq); idx -= b * (ATT_H * nq);
    const int h  = idx / nq;
    const int l0 = (idx % nq) * 64;

#pragma unroll
    for (int i = 0; i < 8; ++i) {
        const int f  = tid + 128 * i;          // 0..1023 float4s
        const int l  = f >> 4;
        const int e4 = (f & 15) << 2;
        const float4 v = *(const float4*)(V + ((size_t)((b * ATT_L + l0 + l) * ATT_H + h)) * ATT_E + e4);
        t[(e4 + 0) * LDK + l] = (_Float16)v.x;
        t[(e4 + 1) * LDK + l] = (_Float16)v.y;
        t[(e4 + 2) * LDK + l] = (_Float16)v.z;
        t[(e4 + 3) * LDK + l] = (_Float16)v.w;
    }
    __syncthreads();
#pragma unroll
    for (int i = 0; i < 4; ++i) {
        const int f  = tid + 128 * i;          // 0..511 8-half chunks
        const int e  = f >> 3;
        const int c8 = (f & 7) << 3;
        v8h o = *(const v8h*)&t[e * LDK + c8];
        *(v8h*)(Vt + ((size_t)((b * ATT_H + h) * ATT_E + e)) * ATT_L + l0 + c8) = o;
    }
}

// ---------------- main attention kernel ----------------

__device__ __forceinline__ void stage_tile(const _Float16* __restrict__ kh,
                                           const _Float16* __restrict__ vt,
                                           _Float16* sKbuf, _Float16* sVbuf,
                                           int b, int h, int k0, int tid)
{
#pragma unroll
    for (int i = 0; i < 4; ++i) {
        const int f   = tid + 128 * i;         // 0..511 16B chunks
        const int row = f >> 3;
        const int c8  = (f & 7) << 3;
        const _Float16* gk = kh + ((size_t)((b * ATT_L + k0 + row) * ATT_H + h)) * ATT_E + c8;
        async_copy16(gk, sKbuf + row * LDK + c8);
        const _Float16* gv = vt + ((size_t)((b * ATT_H + h) * ATT_E + row)) * ATT_L + k0 + c8;
        async_copy16(gv, sVbuf + row * LDK + c8);
    }
}

__global__ void __launch_bounds__(128)
fa_causal_kernel(const _Float16* __restrict__ Qh, const _Float16* __restrict__ Kh,
                 const _Float16* __restrict__ Vt, float* __restrict__ O)
{
    __shared__ _Float16 sK[2][64 * LDK];     // [key][e]
    __shared__ _Float16 sV[2][64 * LDK];     // [e][key]
    __shared__ _Float16 sP[4][16 * LDK];     // per-wave P scratch [row][key]

    const int tid  = threadIdx.x;
    const int w    = tid >> 5;
    const int lane = tid & 31;
    const int nl   = lane & 15;
    const int g    = lane >> 4;

    const int nq = ATT_L / 64;
    int idx = blockIdx.x;
    const int b  = idx / (ATT_H * nq); idx -= b * (ATT_H * nq);
    const int h  = idx / nq;
    const int qt = idx % nq;
    const int qbase = qt * 64;

    // Q fragments (A-layout 16x32 f16), already scaled in pre-pass
    v16h aq[2];
    {
        const _Float16* qrow = Qh + ((size_t)((b * ATT_L + qbase + w * 16 + nl) * ATT_H + h)) * ATT_E;
#pragma unroll
        for (int f = 0; f < 2; ++f) {
            const int o1 = f * 32 + g * 8;
            aq[f] = lds_read_2x8(qrow + o1, qrow + o1 + 16);  // plain b128 global loads
        }
    }

    v16h vones;
#pragma unroll
    for (int i = 0; i < 16; ++i) vones[i] = (_Float16)1.0f;

    v8f oacc[4], lacc;
#pragma unroll
    for (int j = 0; j < 4; ++j)
#pragma unroll
        for (int r = 0; r < 8; ++r) oacc[j][r] = 0.0f;
#pragma unroll
    for (int r = 0; r < 8; ++r) lacc[r] = 0.0f;

    float mrow[8];
#pragma unroll
    for (int r = 0; r < 8; ++r) mrow[r] = -3.0e38f;

    const int ktiles = qt + 1;

    // prologue: stage tile 0
    stage_tile(Kh, Vt, sK[0], sV[0], b, h, 0, tid);
    wait_async0();
    __syncthreads();

    for (int kt = 0; kt < ktiles; ++kt) {
        const int k0  = kt * 64;
        const int cur = kt & 1;

        // overlap: kick off next tile's async copies before computing
        if (kt + 1 < ktiles)
            stage_tile(Kh, Vt, sK[cur ^ 1], sV[cur ^ 1], b, h, k0 + 64, tid);

        // ---- scores: S = (Q*scale) x K^T, 16x64 per wave ----
        v8f c[4];
#pragma unroll
        for (int n = 0; n < 4; ++n) {
            const _Float16* kp = &sK[cur][(n * 16 + nl) * LDK];
            v16h b0 = lds_read_2x8(kp + g * 8, kp + 16 + g * 8);         // e = 0..31
            v16h b1 = lds_read_2x8(kp + 32 + g * 8, kp + 48 + g * 8);    // e = 32..63
            v8f z = {0.f, 0.f, 0.f, 0.f, 0.f, 0.f, 0.f, 0.f};
            c[n] = wmma_f16(aq[0], b0, z);
            c[n] = wmma_f16(aq[1], b1, c[n]);
        }

        // ---- causal mask on the diagonal tile ----
        if (kt == qt) {
#pragma unroll
            for (int n = 0; n < 4; ++n) {
                const int kabs = k0 + n * 16 + nl;
#pragma unroll
                for (int r = 0; r < 8; ++r) {
                    const int qabs = qbase + w * 16 + g * 8 + r;
                    if (kabs > qabs) c[n][r] = -3.0e38f;
                }
            }
        }

        // ---- online softmax: row max (permlane butterflies) + exp ----
#pragma unroll
        for (int r = 0; r < 8; ++r) {
            float t = fmaxf(fmaxf(c[0][r], c[1][r]), fmaxf(c[2][r], c[3][r]));
            t = xor16_max(t);
            const float mnew  = fmaxf(mrow[r], t);
            const float alpha = __expf(mrow[r] - mnew);
            mrow[r] = mnew;
            lacc[r] *= alpha;
#pragma unroll
            for (int j = 0; j < 4; ++j) oacc[j][r] *= alpha;
#pragma unroll
            for (int n = 0; n < 4; ++n)
                c[n][r] = __expf(c[n][r] - mnew);
        }

        // ---- P: C-layout -> LDS -> A-layout (per-wave scratch, DS in-order) ----
        _Float16* pw = &sP[w][0];
#pragma unroll
        for (int n = 0; n < 4; ++n)
#pragma unroll
            for (int r = 0; r < 8; ++r)
                pw[(g * 8 + r) * LDK + n * 16 + nl] = (_Float16)c[n][r];

        asm volatile("s_wait_dscnt 0" ::: "memory");

        v16h pa0 = lds_read_2x8(&pw[nl * LDK + g * 8],      &pw[nl * LDK + 16 + g * 8]);
        v16h pa1 = lds_read_2x8(&pw[nl * LDK + 32 + g * 8], &pw[nl * LDK + 48 + g * 8]);

        // ---- O += P x V ; l += P x 1 (row-sum as an extra WMMA column) ----
#pragma unroll
        for (int j = 0; j < 4; ++j) {
            const _Float16* vp = &sV[cur][(j * 16 + nl) * LDK];
            v16h vb0 = lds_read_2x8(vp + g * 8, vp + 16 + g * 8);
            v16h vb1 = lds_read_2x8(vp + 32 + g * 8, vp + 48 + g * 8);
            oacc[j] = wmma_f16(pa0, vb0, oacc[j]);
            oacc[j] = wmma_f16(pa1, vb1, oacc[j]);
        }
        lacc = wmma_f16(pa0, vones, lacc);
        lacc = wmma_f16(pa1, vones, lacc);

        // next tile's copies must be complete in *all* waves before next round
        if (kt + 1 < ktiles) wait_async0();
        __syncthreads();
    }

    // ---- finalize: O = acc / l ----
#pragma unroll
    for (int r = 0; r < 8; ++r) {
        const float inv = 1.0f / lacc[r];
        const int qa = qbase + w * 16 + g * 8 + r;
        float* orow = O + ((size_t)((b * ATT_L + qa) * ATT_H + h)) * ATT_E;
#pragma unroll
        for (int j = 0; j < 4; ++j)
            orow[j * 16 + nl] = oacc[j][r] * inv;
    }
}

// ---------------- launch ----------------

extern "C" void kernel_launch(void* const* d_in, const int* in_sizes, int n_in,
                              void* d_out, int out_size, void* d_ws, size_t ws_size,
                              hipStream_t stream) {
    const float* q = (const float*)d_in[0];
    const float* k = (const float*)d_in[1];
    const float* v = (const float*)d_in[2];
    float* out = (float*)d_out;

    const size_t nelem = (size_t)ATT_B * ATT_L * ATT_H * ATT_E;   // 4,194,304
    _Float16* Qh = (_Float16*)d_ws;
    _Float16* Kh = Qh + nelem;
    _Float16* Vt = Kh + nelem;

    // pre-pass: convert + transpose
    cvt_qk_kernel<<<dim3((unsigned)(nelem / 4 / 2 / 256)), dim3(256), 0, stream>>>(q, k, Qh, Kh);
    const int ntiles = ATT_B * ATT_H * (ATT_L / 64);              // 1024
    transpose_v_kernel<<<dim3(ntiles), dim3(128), 0, stream>>>(v, Vt);

    // main attention
    fa_causal_kernel<<<dim3(ntiles), dim3(128), 0, stream>>>(Qh, Kh, Vt, out);
}
